// MultiHeadAttention_64879775973985
// MI455X (gfx1250) — matrix operand order repaired and checked by reading, not compiled, not run on hardware
//
#include <hip/hip_runtime.h>
#include <cstdint>
#include <cstddef>

// ---- problem constants -------------------------------------------------
#define Bb   16
#define Tt   1024
#define Cc   1024
#define Hh   16
#define KVH  4
#define Dd   64
#define SCALE_ 0.125f   // 1/sqrt(64)

typedef __bf16 bf16;
typedef __attribute__((ext_vector_type(16))) __bf16 v16bf;
typedef __attribute__((ext_vector_type(8)))  float  v8f;

union Frag {
    v16bf v;
    bf16  h[16];
    unsigned int u[8];
    uint4 q[2];
};

#if defined(__gfx1250__) && __has_builtin(__builtin_amdgcn_tensor_load_to_lds) && __has_builtin(__builtin_amdgcn_s_wait_tensorcnt)
#define USE_TDM 1
#else
#define USE_TDM 0
#endif

#if USE_TDM
typedef unsigned int u32x4 __attribute__((ext_vector_type(4)));
typedef int          i32x4 __attribute__((ext_vector_type(4)));
typedef int          i32x8 __attribute__((ext_vector_type(8)));

// Issue one TDM load of a tileRows x 32-col bf16 tile (row stride = rowLen
// elements) from global into LDS at byte offset ldsAddr. D# per ISA 8.3/8.4.
__device__ __forceinline__ void tdm_load_tile(const bf16* gsrc, unsigned ldsAddr,
                                              int rowLen, int nRowsTensor,
                                              int tileRows) {
    unsigned long long ga = (unsigned long long)(uintptr_t)gsrc;
    u32x4 g0;
    g0[0] = 1u;                                   // count=1, user descriptor
    g0[1] = ldsAddr;                              // lds_addr (bytes)
    g0[2] = (unsigned)(ga & 0xffffffffu);         // global_addr[31:0]
    g0[3] = (unsigned)((ga >> 32) & 0x01ffffffu)  // global_addr[56:32]
          | (2u << 30);                           // type=2 (image)
    i32x8 g1;
    g1[0] = 0x00010000;                           // data_size=1 (2 bytes)
    g1[1] = (int)((unsigned)(rowLen & 0xffff) << 16);            // tensor_dim0 lo
    g1[2] = (int)(((unsigned)rowLen >> 16) & 0xffffu)
          | (int)((unsigned)(nRowsTensor & 0xffff) << 16);       // dim0 hi | dim1 lo
    g1[3] = (int)(((unsigned)nRowsTensor >> 16) & 0xffffu)
          | (int)(32u << 16);                                     // dim1 hi | tile_dim0=32
    g1[4] = tileRows;                             // tile_dim1, tile_dim2=0
    g1[5] = rowLen;                               // tensor_dim0_stride[31:0]
    g1[6] = 0;                                    // stride hi, dim1_stride lo
    g1[7] = 0;
    i32x4 z4 = {0, 0, 0, 0};
#if __clang_major__ >= 23
    i32x8 z8 = {0, 0, 0, 0, 0, 0, 0, 0};
    __builtin_amdgcn_tensor_load_to_lds(g0, g1, z4, z4, z8, 0);
#else
    __builtin_amdgcn_tensor_load_to_lds(g0, g1, z4, z4, 0);
#endif
}
#endif  // USE_TDM

// ---- fp32 -> bf16 convert ---------------------------------------------
__global__ __launch_bounds__(256) void cvt_kernel(const float* __restrict__ src,
                                                  bf16* __restrict__ dst, int n) {
    int i = blockIdx.x * blockDim.x + threadIdx.x;
    int stride = gridDim.x * blockDim.x;
    for (; i < n; i += stride) dst[i] = (bf16)src[i];
}

// ---- fp32 [K,N] -> bf16 transposed [N,K] (32x32 LDS tile) --------------
__global__ __launch_bounds__(256) void cvtT_kernel(const float* __restrict__ src,
                                                   bf16* __restrict__ dst,
                                                   int K, int N) {
    __shared__ float tile[32][33];
    const int kb = blockIdx.y * 32, nb = blockIdx.x * 32;
    const int tx = threadIdx.x & 31, ty = threadIdx.x >> 5;   // 8 rows/pass
#pragma unroll
    for (int i = ty; i < 32; i += 8)
        tile[i][tx] = src[(size_t)(kb + i) * N + nb + tx];
    __syncthreads();
#pragma unroll
    for (int i = ty; i < 32; i += 8)
        dst[(size_t)(nb + i) * K + kb + tx] = (bf16)tile[tx][i];
}

// ---- tiled WMMA GEMM: C[M,N] = A[M,K](bf16) * WT[N,K](bf16), fp32 out --
// block = 256 threads = 8 waves; tile 128(M) x 64(N), BK=32.
// wave w computes rows w*16..+15 over all 64 N: 4 WMMA per k-step sharing
// one A fragment. Both LDS tiles k-contiguous: A frags 8x b32 (merged to
// b128 by the compiler), B frags 2x b128 each.
__global__ __launch_bounds__(256) void gemm_bf16_kernel(
    const bf16* __restrict__ A, const bf16* __restrict__ WT,
    float* __restrict__ Cout, int M, int N, int K) {
    __shared__ __attribute__((aligned(16))) bf16 As[2][128 * 32];
    __shared__ __attribute__((aligned(16))) bf16 Bst[2][64 * 32];

    const int tid  = threadIdx.x;
    const int lane = tid & 31;
    const int wave = tid >> 5;          // 0..7 = M strip
    const int hi   = lane >> 4;
    const int l15  = lane & 15;
    const int mBase = blockIdx.y * 128;
    const int nBase = blockIdx.x * 64;

    v8f acc[4] = {};
    const int nIter = K / 32;

#if USE_TDM
    const unsigned ldsA0 = (unsigned)(uintptr_t)&As[0][0];
    const unsigned ldsA1 = (unsigned)(uintptr_t)&As[1][0];
    const unsigned ldsB0 = (unsigned)(uintptr_t)&Bst[0][0];
    const unsigned ldsB1 = (unsigned)(uintptr_t)&Bst[1][0];
    if (wave == 0) {
        tdm_load_tile(A  + (size_t)mBase * K, ldsA0, K, M, 128);
        tdm_load_tile(WT + (size_t)nBase * K, ldsB0, K, N, 64);
    }
#else
    const int rA = tid >> 1, cA = (tid & 1) * 16;  // 128 rows x 32 cols, 16 e/thr
    const int rB = tid >> 2, cB = (tid & 3) * 8;   // 64 rows x 32 cols,  8 e/thr
#endif

    for (int it = 0; it < nIter; ++it) {
        const int cur = it & 1;
#if USE_TDM
        if (wave == 0) __builtin_amdgcn_s_wait_tensorcnt(0);
        __syncthreads();                    // staged tile visible; old buffer free
        if (it + 1 < nIter && wave == 0) {
            const int k0n = (it + 1) * 32;
            tdm_load_tile(A  + (size_t)mBase * K + k0n, cur ? ldsA0 : ldsA1, K, M, 128);
            tdm_load_tile(WT + (size_t)nBase * K + k0n, cur ? ldsB0 : ldsB1, K, N, 64);
        }
#else
        const int k0 = it * 32;
        __syncthreads();
        uint4 a0 = *reinterpret_cast<const uint4*>(A  + (size_t)(mBase + rA) * K + k0 + cA);
        uint4 a1 = *reinterpret_cast<const uint4*>(A  + (size_t)(mBase + rA) * K + k0 + cA + 8);
        uint4 bv = *reinterpret_cast<const uint4*>(WT + (size_t)(nBase + rB) * K + k0 + cB);
        reinterpret_cast<uint4*>(As[0])[tid * 2]     = a0;
        reinterpret_cast<uint4*>(As[0])[tid * 2 + 1] = a1;
        reinterpret_cast<uint4*>(Bst[0])[tid]        = bv;
        __syncthreads();
#endif
        const bf16* Asb = As[USE_TDM ? cur : 0];
        const bf16* Bsb = Bst[USE_TDM ? cur : 0];

        // A fragment, 16x32 bf16: lane l15 = row M; vgpr p holds K pair
        Frag af;
#pragma unroll
        for (int p = 0; p < 8; ++p) {
            int kk = ((p >= 4) ? 16 : 0) + hi * 8 + (p & 3) * 2;
            af.u[p] = *reinterpret_cast<const unsigned int*>(&Asb[(wave * 16 + l15) * 32 + kk]);
        }
        // B fragments, 32x16 bf16: lane l15 = col N; element e -> K = (e>=8?16:0)+hi*8+(e&7)
#pragma unroll
        for (int t = 0; t < 4; ++t) {
            Frag bt;
            bt.q[0] = *reinterpret_cast<const uint4*>(&Bsb[(t * 16 + l15) * 32 + hi * 8]);
            bt.q[1] = *reinterpret_cast<const uint4*>(&Bsb[(t * 16 + l15) * 32 + hi * 8 + 16]);
            acc[t] = __builtin_amdgcn_wmma_f32_16x16x32_bf16(false, af.v, false, bt.v, (short)0, acc[t], false, false);
        }
    }

#pragma unroll
    for (int r = 0; r < 8; ++r) {
        int row = mBase + wave * 16 + hi * 8 + r;
#pragma unroll
        for (int t = 0; t < 4; ++t)
            Cout[(size_t)row * N + nBase + t * 16 + l15] = acc[t][r];
    }
}

// ---- fused RoPE + RMSNorm epilogue (+ V transpose to d-major) ----------
// one wave per (token, head-job). jobs: 0..15 q heads, 16..19 k heads, 20..23 v heads
__global__ __launch_bounds__(256) void rope_norm_kernel(
    const float* __restrict__ qf, const float* __restrict__ kf, const float* __restrict__ vf,
    const float* __restrict__ cosT, const float* __restrict__ sinT,
    const float* __restrict__ qw, const float* __restrict__ kw,
    bf16* __restrict__ qb, bf16* __restrict__ kb, bf16* __restrict__ vbt) {
    const int lane = threadIdx.x & 31;
    const int wave = threadIdx.x >> 5;
    const long long task = (long long)blockIdx.x * 8 + wave;
    const int token = (int)(task / 24);
    const int job   = (int)(task % 24);
    const int t = token % Tt;
    const int b = token / Tt;

    if (job < 20) {
        const float* src; bf16* dst; const float* w; int nh, head;
        if (job < 16) { src = qf; dst = qb; w = qw; nh = Hh;  head = job; }
        else          { src = kf; dst = kb; w = kw; nh = KVH; head = job - 16; }
        const float* p = src + ((size_t)token * nh + head) * Dd;
        float a  = p[lane];
        float b2 = p[lane + 32];
        float c  = cosT[t * 32 + lane];
        float s  = sinT[t * 32 + lane];
        float r1 = a * c + b2 * s;
        float r2 = -a * s + b2 * c;
        float ss = r1 * r1 + r2 * r2;
#pragma unroll
        for (int off = 16; off > 0; off >>= 1) ss += __shfl_xor(ss, off, 32);
        float rms = rsqrtf(ss * (1.0f / 64.0f) + 1e-6f);
        bf16* o = dst + ((size_t)token * nh + head) * Dd;
        o[lane]      = (bf16)(r1 * rms * w[lane]);
        o[lane + 32] = (bf16)(r2 * rms * w[lane + 32]);
    } else {
        int g = job - 20;
        const float* p = vf + ((size_t)token * KVH + g) * Dd;
        bf16* o = vbt + ((size_t)(b * KVH + g) * Dd) * (size_t)Tt;
        o[(size_t)lane * Tt + t]        = (bf16)p[lane];
        o[(size_t)(lane + 32) * Tt + t] = (bf16)p[lane + 32];
    }
}

// ---- flash-style WMMA attention ---------------------------------------
// one wave per (b, head, 16-row query tile); 8 waves/block share same t0.
// block-causal mask => key tiles fully live up to kend = (t0/32+1)*32.
__global__ __launch_bounds__(256) void attn_kernel(
    const bf16* __restrict__ qb, const bf16* __restrict__ kb,
    const bf16* __restrict__ vbt, bf16* __restrict__ yb) {
    __shared__ __attribute__((aligned(16))) bf16 plds[8 * 16 * 32];  // per-wave 16x32 P tile

    const int tid  = threadIdx.x;
    const int lane = tid & 31;
    const int wave = tid >> 5;
    const int hi   = lane >> 4;
    const int l15  = lane & 15;
    const int wg   = blockIdx.x * 8 + wave;
    const int h    = wg % Hh;
    const int tile = (wg / Hh) % (Tt / 16);
    const int b    = wg / (Hh * (Tt / 16));
    const int g    = h >> 2;           // kv group
    const int t0   = tile * 16;
    bf16* pl = plds + wave * 512;

    // Q A-fragments: rows t0..t0+15, head-dim split into two K=32 chunks
    const bf16* qp = qb + ((size_t)(b * Tt + t0 + l15) * Hh + h) * (size_t)Dd;
    Frag aq0, aq1;
#pragma unroll
    for (int p = 0; p < 8; ++p) {
        int kk = ((p >= 4) ? 16 : 0) + hi * 8 + (p & 3) * 2;
        aq0.u[p] = *reinterpret_cast<const unsigned int*>(qp + kk);
        aq1.u[p] = *reinterpret_cast<const unsigned int*>(qp + 32 + kk);
    }

    v8f o0 = {}, o1 = {}, o2 = {}, o3 = {};
    float mrow[8], lrow[8];
#pragma unroll
    for (int r = 0; r < 8; ++r) { mrow[r] = -1e30f; lrow[r] = 0.0f; }

    const int kend = ((t0 >> 5) + 1) << 5;
    for (int s0 = 0; s0 < kend; s0 += 32) {
        if (s0 + 32 < kend) {   // prefetch next K chunk (global_prefetch_b8)
            __builtin_prefetch(kb + ((size_t)(b * Tt + s0 + 32 + l15) * KVH + g) * (size_t)Dd, 0, 0);
        }
        // ---- scores: two 16-key tiles, each = 2 WMMA over head-dim -----
        v8f sA, sB;
#pragma unroll
        for (int nt = 0; nt < 2; ++nt) {
            const bf16* kp = kb + ((size_t)(b * Tt + s0 + nt * 16 + l15) * KVH + g) * (size_t)Dd + hi * 8;
            Frag bk0, bk1;                    // B = K^T: K-dim = d, N = key
            bk0.q[0] = *reinterpret_cast<const uint4*>(kp);
            bk0.q[1] = *reinterpret_cast<const uint4*>(kp + 16);
            bk1.q[0] = *reinterpret_cast<const uint4*>(kp + 32);
            bk1.q[1] = *reinterpret_cast<const uint4*>(kp + 48);
            v8f z = {};
            v8f t = __builtin_amdgcn_wmma_f32_16x16x32_bf16(false, aq0.v, false, bk0.v, (short)0, z, false, false);
            t     = __builtin_amdgcn_wmma_f32_16x16x32_bf16(false, aq1.v, false, bk1.v, (short)0, t, false, false);
            if (nt == 0) sA = t; else sB = t;
        }

        // ---- online softmax over 32 keys -------------------------------
        float mc[8], p0[8], p1[8], alpha[8], rs[8];
#pragma unroll
        for (int r = 0; r < 8; ++r) {
            sA[r] *= SCALE_; sB[r] *= SCALE_;
            mc[r] = fmaxf(sA[r], sB[r]);
        }
#pragma unroll
        for (int off = 1; off < 16; off <<= 1)
#pragma unroll
            for (int r = 0; r < 8; ++r)
                mc[r] = fmaxf(mc[r], __shfl_xor(mc[r], off, 32));
#pragma unroll
        for (int r = 0; r < 8; ++r) {
            float mn = fmaxf(mrow[r], mc[r]);
            alpha[r] = __expf(mrow[r] - mn);
            mrow[r]  = mn;
            p0[r] = __expf(sA[r] - mn);
            p1[r] = __expf(sB[r] - mn);
            rs[r] = p0[r] + p1[r];
        }
#pragma unroll
        for (int off = 1; off < 16; off <<= 1)
#pragma unroll
            for (int r = 0; r < 8; ++r)
                rs[r] += __shfl_xor(rs[r], off, 32);
#pragma unroll
        for (int r = 0; r < 8; ++r) {
            lrow[r] = lrow[r] * alpha[r] + rs[r];
            o0[r] *= alpha[r]; o1[r] *= alpha[r]; o2[r] *= alpha[r]; o3[r] *= alpha[r];
        }

        // ---- re-layout P (C layout) -> A fragment via per-wave LDS -----
#pragma unroll
        for (int r = 0; r < 8; ++r) {
            int M = r + hi * 8;
            pl[M * 32 + l15]      = (bf16)p0[r];
            pl[M * 32 + 16 + l15] = (bf16)p1[r];
        }
        Frag pa;
#pragma unroll
        for (int p = 0; p < 8; ++p) {
            int kk = ((p >= 4) ? 16 : 0) + hi * 8 + (p & 3) * 2;
            pa.u[p] = *reinterpret_cast<const unsigned int*>(&pl[l15 * 32 + kk]);
        }

        // ---- P(16x32) @ V(32x64): 4 WMMA, V is d-major -----------------
#pragma unroll
        for (int dt = 0; dt < 4; ++dt) {
            const bf16* vp = vbt + ((size_t)(b * KVH + g) * Dd + dt * 16 + l15) * (size_t)Tt + s0 + hi * 8;
            Frag bv;
            bv.q[0] = *reinterpret_cast<const uint4*>(vp);
            bv.q[1] = *reinterpret_cast<const uint4*>(vp + 16);
            v8f& od = (dt == 0 ? o0 : dt == 1 ? o1 : dt == 2 ? o2 : o3);
            od = __builtin_amdgcn_wmma_f32_16x16x32_bf16(false, pa.v, false, bv.v, (short)0, od, false, false);
        }
    }

    // ---- normalize and store y (bf16, [B,T,H,D]) ------------------------
#pragma unroll
    for (int r = 0; r < 8; ++r) {
        float inv = 1.0f / lrow[r];
        int trow = t0 + r + hi * 8;
        bf16* op = yb + ((size_t)(b * Tt + trow) * Hh + h) * (size_t)Dd;
        op[l15]      = (bf16)(o0[r] * inv);
        op[16 + l15] = (bf16)(o1[r] * inv);
        op[32 + l15] = (bf16)(o2[r] * inv);
        op[48 + l15] = (bf16)(o3[r] * inv);
    }
}

// ---- workspace layout (bytes) ------------------------------------------
#define OFF_XB   ((size_t)0)                   // 16M bf16   = 33554432
#define OFF_WQB  ((size_t)33554432)            // 1M  bf16 (transposed N x K)
#define OFF_WKB  ((size_t)35651584)            // 256K bf16 (transposed)
#define OFF_WVB  ((size_t)36175872)            // (transposed)
#define OFF_WOB  ((size_t)36700160)            // 1M bf16 (transposed)
#define OFF_QF   ((size_t)38797312)            // 16M f32
#define OFF_KF   ((size_t)105906176)           // 4M f32
#define OFF_VF   ((size_t)122683392)           // 4M f32
#define OFF_QB   ((size_t)139460608)           // 16M bf16
#define OFF_KB   ((size_t)173015040)           // 4M bf16
#define OFF_VBT  ((size_t)181403648)           // 4M bf16 (d-major)
#define OFF_YB   ((size_t)189792256)           // 16M bf16

extern "C" void kernel_launch(void* const* d_in, const int* in_sizes, int n_in,
                              void* d_out, int out_size, void* d_ws, size_t ws_size,
                              hipStream_t stream) {
    const float* x    = (const float*)d_in[0];
    const float* cosT = (const float*)d_in[1];
    const float* sinT = (const float*)d_in[2];
    // d_in[3] attn_mask: block-causal structure handled analytically
    const float* Wq = (const float*)d_in[4];
    const float* Wk = (const float*)d_in[5];
    const float* Wv = (const float*)d_in[6];
    const float* Wo = (const float*)d_in[7];
    const float* qw = (const float*)d_in[8];
    const float* kw = (const float*)d_in[9];
    float* out = (float*)d_out;

    char* ws = (char*)d_ws;
    bf16*  xb  = (bf16*)(ws + OFF_XB);
    bf16*  WqT = (bf16*)(ws + OFF_WQB);
    bf16*  WkT = (bf16*)(ws + OFF_WKB);
    bf16*  WvT = (bf16*)(ws + OFF_WVB);
    bf16*  WoT = (bf16*)(ws + OFF_WOB);
    float* qf  = (float*)(ws + OFF_QF);
    float* kf  = (float*)(ws + OFF_KF);
    float* vf  = (float*)(ws + OFF_VF);
    bf16*  qb  = (bf16*)(ws + OFF_QB);
    bf16*  kb  = (bf16*)(ws + OFF_KB);
    bf16*  vbt = (bf16*)(ws + OFF_VBT);
    bf16*  yb  = (bf16*)(ws + OFF_YB);

    const int M = Bb * Tt;   // 16384

    // 1) bf16 conversions (weights transposed to N x K)
    cvt_kernel<<<4096, 256, 0, stream>>>(x, xb, Bb * Tt * Cc);
    cvtT_kernel<<<dim3(1024 / 32, 1024 / 32), 256, 0, stream>>>(Wq, WqT, 1024, 1024);
    cvtT_kernel<<<dim3(256  / 32, 1024 / 32), 256, 0, stream>>>(Wk, WkT, 1024, 256);
    cvtT_kernel<<<dim3(256  / 32, 1024 / 32), 256, 0, stream>>>(Wv, WvT, 1024, 256);
    cvtT_kernel<<<dim3(1024 / 32, 1024 / 32), 256, 0, stream>>>(Wo, WoT, 1024, 1024);

    // 2) QKV projections (WMMA GEMM, TDM-staged LDS, fp32 out)
    gemm_bf16_kernel<<<dim3(1024 / 64, M / 128), 256, 0, stream>>>(xb, WqT, qf, M, 1024, 1024);
    gemm_bf16_kernel<<<dim3(256  / 64, M / 128), 256, 0, stream>>>(xb, WkT, kf, M, 256,  1024);
    gemm_bf16_kernel<<<dim3(256  / 64, M / 128), 256, 0, stream>>>(xb, WvT, vf, M, 256,  1024);

    // 3) RoPE + RMSNorm (+ V transpose), bf16 out
    rope_norm_kernel<<<(M * 24) / 8, 256, 0, stream>>>(qf, kf, vf, cosT, sinT, qw, kw, qb, kb, vbt);

    // 4) flash attention (WMMA), bf16 y
    attn_kernel<<<(Bb * Hh * (Tt / 16)) / 8, 256, 0, stream>>>(qb, kb, vbt, yb);

    // 5) output projection, fp32 out
    gemm_bf16_kernel<<<dim3(1024 / 64, M / 128), 256, 0, stream>>>(yb, WoT, out, M, 1024, 1024);
}
